// BertSelfAttentionNaiveImpl_45629732552747
// MI455X (gfx1250) — matrix operand order repaired and checked by reading, not compiled, not run on hardware
//
#include <hip/hip_runtime.h>
#include <hip/hip_bf16.h>

typedef __attribute__((ext_vector_type(16))) __bf16 v16bf;
typedef __attribute__((ext_vector_type(8)))  float  v8f;

#define NH 12
#define SEQ 1024
#define HID 768
#define HD 64
#define RWIN 129   // 2*64+1

// ---------------- CDNA5 async copy helpers ----------------

__device__ static inline void async_load16(unsigned lds_off, const void* gaddr) {
    // GLOBAL_LOAD_ASYNC_TO_LDS_B128, GV mode (64-bit vaddr), tracked by ASYNCcnt
    asm volatile("global_load_async_to_lds_b128 %0, %1, off"
                 :: "v"(lds_off), "v"((unsigned long long)(uintptr_t)gaddr)
                 : "memory");
}
#define WAIT_ASYNC_8() asm volatile("s_wait_asynccnt 0x8" ::: "memory")
#define WAIT_ASYNC_0() asm volatile("s_wait_asynccnt 0x0" ::: "memory")

__device__ static inline unsigned lds_offset(const void* p) {
    // generic pointer to LDS: low 32 bits are the LDS byte offset
    return (unsigned)(uintptr_t)p;
}

// ---------------- fragment loaders (CDNA5 WMMA bf16 layouts) ----------------

// A-matrix 16x32 bf16, row-major source
__device__ static inline v16bf load_a_frag(const __bf16* base, int ld, int lane) {
    int m  = lane & 15;
    int kb = (lane < 16) ? 0 : 16;
    const __bf16* row = base + m * ld;
    v16bf a;
#pragma unroll
    for (int i = 0; i < 4; ++i) {
        int k = kb + 2 * i;
        a[2 * i] = row[k]; a[2 * i + 1] = row[k + 1];
    }
#pragma unroll
    for (int i = 0; i < 4; ++i) {
        int k = kb + 8 + 2 * i;
        a[8 + 2 * i] = row[k]; a[9 + 2 * i] = row[k + 1];
    }
    return a;
}

// B-matrix 32x16 bf16 where element (kk,n) = base[n*ld + kk]
__device__ static inline v16bf load_b_row(const __bf16* base, int ld, int lane) {
    int n  = lane & 15;
    int kb = (lane < 16) ? 0 : 16;
    const __bf16* row = base + n * ld + kb;
    v16bf b;
#pragma unroll
    for (int i = 0; i < 8; ++i) {
        b[2 * i] = row[2 * i]; b[2 * i + 1] = row[2 * i + 1];
    }
    return b;
}

// ---------------- conversion kernels ----------------

__global__ void k_f32_to_bf16(const float* __restrict__ src, __bf16* __restrict__ dst, int n) {
    int i = blockIdx.x * 256 + threadIdx.x;
    if (i < n) dst[i] = (__bf16)src[i];
}

// Wt[n][k] = W[k][n], 768x768
__global__ void k_transpose_bf16(const float* __restrict__ W, __bf16* __restrict__ Wt) {
    int i = blockIdx.x * 256 + threadIdx.x;
    if (i < HID * HID) {
        int n = i / HID, k = i % HID;
        Wt[i] = (__bf16)W[k * HID + n];
    }
}

// Vt[bh][d][j] = V[b][j][h*64+d]  (per-head transposed V for contiguous PV B-frags)
__global__ void k_transpose_v(const __bf16* __restrict__ Vb, __bf16* __restrict__ Vt) {
    int i = blockIdx.x * 256 + threadIdx.x;
    if (i < 4 * NH * HD * SEQ) {
        int j  = i & (SEQ - 1);
        int r  = i >> 10;
        int d  = r & (HD - 1);
        int bh = r >> 6;
        int b = bh / NH, h = bh % NH;
        Vt[i] = Vb[((size_t)(b * SEQ + j)) * HID + h * HD + d];
    }
}

// ---------------- QKV GEMM: Out[4096x768] = X @ W + bias (bf16, f32 acc, async double-buffer) ----------------

__global__ void k_gemm_qkv(const __bf16* __restrict__ Xb, const __bf16* __restrict__ Wt,
                           const float* __restrict__ bias, __bf16* __restrict__ Out) {
    __shared__ __align__(16) __bf16 Xs[2][64][64];
    __shared__ __align__(16) __bf16 Ws[2][64][64];

    const int tid  = threadIdx.x;
    const int wave = tid >> 5;
    const int lane = tid & 31;
    const int m0 = blockIdx.x * 64;
    const int n0 = blockIdx.y * 64;

    const unsigned xs0 = lds_offset(&Xs[0][0][0]);
    const unsigned ws0 = lds_offset(&Ws[0][0][0]);

    v8f acc[4];
#pragma unroll
    for (int t = 0; t < 4; ++t) acc[t] = (v8f){0.f,0.f,0.f,0.f,0.f,0.f,0.f,0.f};

    // prologue: stage tile k0=0 into buffer 0 (4 chunks each of X and W per thread)
#pragma unroll
    for (int i = 0; i < 4; ++i) {
        int c0 = tid + 128 * i;            // 16B chunk id within 64x64 bf16 tile
        int row = c0 >> 3, col = c0 & 7;
        async_load16(xs0 + c0 * 16, Xb + (size_t)(m0 + row) * HID + col * 8);
        async_load16(ws0 + c0 * 16, Wt + (size_t)(n0 + row) * HID + col * 8);
    }

    for (int kt = 0; kt < HID / 64; ++kt) {
        const int cur = kt & 1;
        __syncthreads();                    // previous consumers of 'nxt' buffer done
        if (kt + 1 < HID / 64) {
            const int nxt = cur ^ 1;
            const int k1 = (kt + 1) * 64;
            const unsigned xb = xs0 + nxt * 64 * 64 * 2;
            const unsigned wb = ws0 + nxt * 64 * 64 * 2;
#pragma unroll
            for (int i = 0; i < 4; ++i) {
                int c0 = tid + 128 * i;
                int row = c0 >> 3, col = c0 & 7;
                async_load16(xb + c0 * 16, Xb + (size_t)(m0 + row) * HID + k1 + col * 8);
                async_load16(wb + c0 * 16, Wt + (size_t)(n0 + row) * HID + k1 + col * 8);
            }
            WAIT_ASYNC_8();                 // current buffer resident (in-order)
        } else {
            WAIT_ASYNC_0();
        }
        __syncthreads();                    // publish current buffer to all waves

#pragma unroll
        for (int c = 0; c < 2; ++c) {
            v16bf a = load_a_frag(&Xs[cur][wave * 16][c * 32], 64, lane);
#pragma unroll
            for (int t = 0; t < 4; ++t) {
                v16bf b = load_b_row(&Ws[cur][t * 16][c * 32], 64, lane);
                acc[t] = __builtin_amdgcn_wmma_f32_16x16x32_bf16(false, a, false, b,
                                                                 (short)0, acc[t], false, false);
            }
        }
    }

    const int rbase = m0 + wave * 16 + ((lane < 16) ? 0 : 8);
    const int ln = lane & 15;
#pragma unroll
    for (int t = 0; t < 4; ++t) {
        int col = n0 + t * 16 + ln;
        float bcol = bias[col];
#pragma unroll
        for (int r = 0; r < 8; ++r)
            Out[(size_t)(rbase + r) * HID + col] = (__bf16)(acc[t][r] + bcol);
    }
}

// ---------------- rel-key projection: Prel[bh,i,r] = q_i . E_key[r] ----------------

__global__ void k_relkey(const __bf16* __restrict__ Qb, const float* __restrict__ Ek,
                         float* __restrict__ Prel) {
    __shared__ float qrow[HD];
    int blk = blockIdx.x;              // bh*SEQ + i
    int i   = blk & (SEQ - 1);
    int bh  = blk >> 10;
    int b   = bh / NH, h = bh % NH;
    int t = threadIdx.x;
    if (t < HD) qrow[t] = (float)Qb[((size_t)(b * SEQ + i)) * HID + h * HD + t];
    __syncthreads();
    if (t < RWIN) {
        const float* e = Ek + t * HD;
        float s = 0.f;
#pragma unroll 8
        for (int d = 0; d < HD; ++d) s += qrow[d] * e[d];
        Prel[((size_t)bh * SEQ + i) * RWIN + t] = s;
    }
}

// ---------------- fused streaming attention (flash-style, async double-buffered K/Vt) ----------------

__global__ void k_attention(const __bf16* __restrict__ Qb, const __bf16* __restrict__ Kb,
                            const __bf16* __restrict__ Vt, const float* __restrict__ Prel,
                            const float* __restrict__ mask, const float* __restrict__ Ev,
                            float* __restrict__ out) {
    __shared__ __align__(16) __bf16 Qs[64][64];
    __shared__ __align__(16) __bf16 Ks[2][64][64];
    __shared__ __align__(16) __bf16 VtS[2][64][64];   // [d][j] per-head transposed tile
    __shared__ float  Sbuf[64][64];
    __shared__ __bf16 Pbuf[64][64];
    __shared__ float  CtxPos[64][64];
    __shared__ float  EvS[RWIN][HD];
    __shared__ float  fac[64];
    __shared__ float  linv[64];

    const int tid  = threadIdx.x;
    const int wave = tid >> 5;
    const int lane = tid & 31;
    const int bid = blockIdx.x;
    const int ib  = bid & 15;
    const int bh  = bid >> 4;
    const int h = bh % NH, b = bh / NH;
    const int i0 = ib * 64;

    const __bf16* Kbase  = Kb + ((size_t)b * SEQ) * HID + h * HD;    // row j: + j*HID
    const __bf16* VtBase = Vt + ((size_t)bh * HD) * SEQ;             // row d: + d*SEQ
    const unsigned ks0  = lds_offset(&Ks[0][0][0]);
    const unsigned vts0 = lds_offset(&VtS[0][0][0]);

    // stage Q tile, E_val, zero ctx-pos
    for (int e = tid; e < 64 * 64; e += 128) {
        int r = e >> 6, c = e & 63;
        Qs[r][c] = Qb[((size_t)(b * SEQ + i0 + r)) * HID + h * HD + c];
        CtxPos[r][c] = 0.f;
    }
    for (int e = tid; e < RWIN * HD; e += 128) EvS[e / HD][e % HD] = Ev[e];

    // prologue: async-stage j-block 0 into buffer 0
#pragma unroll
    for (int i = 0; i < 4; ++i) {
        int c0 = tid + 128 * i;
        int row = c0 >> 3, col = c0 & 7;
        async_load16(ks0  + c0 * 16, Kbase  + (size_t)row * HID + col * 8);
        async_load16(vts0 + c0 * 16, VtBase + (size_t)row * SEQ + col * 8);
    }
    __syncthreads();

    // Q A-fragments live in registers for the whole stream
    v16bf aq[2];
#pragma unroll
    for (int c = 0; c < 2; ++c) aq[c] = load_a_frag(&Qs[wave * 16][c * 32], 64, lane);

    v8f acc[4];
#pragma unroll
    for (int t = 0; t < 4; ++t) acc[t] = (v8f){0.f,0.f,0.f,0.f,0.f,0.f,0.f,0.f};

    // two adjacent lanes own one row: row = tid>>1, parity selects 32-column half
    const int srow_id = tid >> 1;
    const int par     = tid & 1;
    float m_run = -__builtin_inff();
    float l_run = 0.f;

    for (int jt = 0; jt < 16; ++jt) {
        const int j0  = jt * 64;
        const int cur = jt & 1;
        __syncthreads();                    // all consumers of 'nxt' buffer + Pbuf done
        if (jt + 1 < 16) {
            const int nxt = cur ^ 1;
            const int j1 = (jt + 1) * 64;
            const unsigned kb = ks0  + nxt * 64 * 64 * 2;
            const unsigned vb = vts0 + nxt * 64 * 64 * 2;
#pragma unroll
            for (int i = 0; i < 4; ++i) {
                int c0 = tid + 128 * i;
                int row = c0 >> 3, col = c0 & 7;
                async_load16(kb + c0 * 16, Kbase  + (size_t)(j1 + row) * HID + col * 8);
                async_load16(vb + c0 * 16, VtBase + (size_t)row * SEQ + j1 + col * 8);
            }
            WAIT_ASYNC_8();
        } else {
            WAIT_ASYNC_0();
        }
        __syncthreads();                    // current K/Vt tiles visible to all waves

        // S = Q K^T  (64x64 f32 tile)
#pragma unroll
        for (int t = 0; t < 4; ++t) {
            v8f cs = (v8f){0.f,0.f,0.f,0.f,0.f,0.f,0.f,0.f};
#pragma unroll
            for (int c = 0; c < 2; ++c) {
                v16bf bk = load_b_row(&Ks[cur][t * 16][c * 32], 64, lane);
                cs = __builtin_amdgcn_wmma_f32_16x16x32_bf16(false, aq[c], false, bk,
                                                             (short)0, cs, false, false);
            }
            int colb = t * 16 + (lane & 15);
            int rowb = wave * 16 + ((lane < 16) ? 0 : 8);
#pragma unroll
            for (int r = 0; r < 8; ++r) Sbuf[rowb + r][colb] = cs[r];
        }
        __syncthreads();

        // online softmax: lane pair per row, shuffle-combined stats
        {
            const int row = srow_id;
            const int ig  = i0 + row;
            float* srow = Sbuf[row];
            const float* prel = Prel + ((size_t)bh * SEQ + ig) * RWIN;
            const int c0 = par * 32;
            float mloc = -__builtin_inff();
            for (int j = c0; j < c0 + 32; ++j) {
                int jg = j0 + j;
                float s = srow[j];
                int dl = jg - ig + 64;
                if ((unsigned)dl < (unsigned)RWIN) s += prel[dl];
                s += (1.0f - mask[b * SEQ + jg]) * -3.402823466e38f;
                srow[j] = s;
                mloc = fmaxf(mloc, s);
            }
            mloc = fmaxf(mloc, __shfl_xor(mloc, 1));
            float mnew = fmaxf(m_run, mloc);
            float f = __expf(m_run - mnew);
            float sum = 0.f;
            for (int j = c0; j < c0 + 32; ++j) {
                float e = __expf(srow[j] - mnew);
                srow[j] = e;
                Pbuf[row][j] = (__bf16)e;
                sum += e;
            }
            sum += __shfl_xor(sum, 1);
            l_run = l_run * f + sum;
            m_run = mnew;
            if (par == 0) fac[row] = f;
            // rescale ctx-pos accumulator (this lane's d-half)
            for (int d = c0; d < c0 + 32; ++d) CtxPos[row][d] *= f;
            // banded E_val accumulation (only blocks intersecting the +-64 band)
            int jlo = (ig - 64 > j0) ? (ig - 64) : j0;
            int jhi = (ig + 64 < j0 + 63) ? (ig + 64) : (j0 + 63);
            for (int jg = jlo; jg <= jhi; ++jg) {
                float e = srow[jg - j0];
                const float* ev = EvS[jg - ig + 64];
#pragma unroll 8
                for (int d = c0; d < c0 + 32; ++d) CtxPos[row][d] += e * ev[d];
            }
        }
        __syncthreads();

        // rescale O accumulators, then O += P V
        float fr[8];
#pragma unroll
        for (int r = 0; r < 8; ++r) fr[r] = fac[wave * 16 + r + ((lane < 16) ? 0 : 8)];
#pragma unroll
        for (int t = 0; t < 4; ++t)
#pragma unroll
            for (int r = 0; r < 8; ++r) acc[t][r] *= fr[r];

#pragma unroll
        for (int c = 0; c < 2; ++c) {
            v16bf ap = load_a_frag(&Pbuf[wave * 16][c * 32], 64, lane);
#pragma unroll
            for (int t = 0; t < 4; ++t) {
                // B(kk=j_local, n=d) = VtS[d][j_local] -> contiguous b128 frag loads
                v16bf bv = load_b_row(&VtS[cur][t * 16][c * 32], 64, lane);
                acc[t] = __builtin_amdgcn_wmma_f32_16x16x32_bf16(false, ap, false, bv,
                                                                 (short)0, acc[t], false, false);
            }
        }
    }

    if (par == 0) linv[srow_id] = 1.0f / l_run;
    __syncthreads();

    const int rowb = wave * 16 + ((lane < 16) ? 0 : 8);
    const int ln = lane & 15;
#pragma unroll
    for (int t = 0; t < 4; ++t) {
        int col = t * 16 + ln;
#pragma unroll
        for (int r = 0; r < 8; ++r) {
            int row = rowb + r;
            int ig = i0 + row;
            float v = (acc[t][r] + CtxPos[row][col]) * linv[row];
            out[((size_t)(b * SEQ + ig)) * HID + h * HD + col] = v;
        }
    }
}

// ---------------- host launch ----------------

extern "C" void kernel_launch(void* const* d_in, const int* in_sizes, int n_in,
                              void* d_out, int out_size, void* d_ws, size_t ws_size,
                              hipStream_t stream) {
    const float* hidden = (const float*)d_in[0];
    const float* amask  = (const float*)d_in[1];
    const float* Wq = (const float*)d_in[2];
    const float* bq = (const float*)d_in[3];
    const float* Wk = (const float*)d_in[4];
    const float* bk = (const float*)d_in[5];
    const float* Wv = (const float*)d_in[6];
    const float* bv = (const float*)d_in[7];
    const float* Ek = (const float*)d_in[8];
    const float* Ev = (const float*)d_in[9];
    float* out = (float*)d_out;

    const int B = 4;
    const size_t ROWS = (size_t)B * SEQ;          // 4096
    const size_t XB_BYTES = ROWS * HID * 2;       // bf16 activations
    const size_t WT_BYTES = (size_t)HID * HID * 2;

    char* w = (char*)d_ws;
    __bf16* Xb  = (__bf16*)w;                w += XB_BYTES;
    __bf16* Wtq = (__bf16*)w;                w += WT_BYTES;
    __bf16* Wtk = (__bf16*)w;                w += WT_BYTES;
    __bf16* Wtv = (__bf16*)w;                w += WT_BYTES;
    __bf16* Qb  = (__bf16*)w;                w += XB_BYTES;
    __bf16* Kb  = (__bf16*)w;                w += XB_BYTES;
    __bf16* Vb  = (__bf16*)w;                w += XB_BYTES;
    __bf16* Vtp = (__bf16*)w;                w += XB_BYTES;   // per-head transposed V
    float*  Prel = (float*)w;                // B*NH*SEQ*RWIN f32

    // 1) convert activations + weights to bf16
    int nelem = (int)(ROWS * HID);
    k_f32_to_bf16<<<(nelem + 255) / 256, 256, 0, stream>>>(hidden, Xb, nelem);
    int nw = HID * HID;
    k_transpose_bf16<<<(nw + 255) / 256, 256, 0, stream>>>(Wq, Wtq);
    k_transpose_bf16<<<(nw + 255) / 256, 256, 0, stream>>>(Wk, Wtk);
    k_transpose_bf16<<<(nw + 255) / 256, 256, 0, stream>>>(Wv, Wtv);

    // 2) QKV projections (WMMA bf16, async double-buffered tiles)
    dim3 ggrid(ROWS / 64, HID / 64);
    k_gemm_qkv<<<ggrid, 128, 0, stream>>>(Xb, Wtq, bq, Qb);
    k_gemm_qkv<<<ggrid, 128, 0, stream>>>(Xb, Wtk, bk, Kb);
    k_gemm_qkv<<<ggrid, 128, 0, stream>>>(Xb, Wtv, bv, Vb);

    // 2b) per-head transpose of V for contiguous PV B-fragments
    k_transpose_v<<<(nelem + 255) / 256, 256, 0, stream>>>(Vb, Vtp);

    // 3) rel-key projections
    k_relkey<<<B * NH * SEQ, 160, 0, stream>>>(Qb, Ek, Prel);

    // 4) fused streaming attention (flash-style, banded rel-pos)
    k_attention<<<B * NH * (SEQ / 64), 128, 0, stream>>>(Qb, Kb, Vtp, Prel, amask, Ev, out);
}
